// SPVCNN152_cls_78503412236512
// MI455X (gfx1250) — matrix operand order repaired and checked by reading, not compiled, not run on hardware
//
#include <hip/hip_runtime.h>
#include <hip/hip_bf16.h>
#include <algorithm>
#include <utility>

typedef __bf16 bf16_t;
typedef unsigned int u32;
typedef __attribute__((ext_vector_type(16))) __bf16 bfvec16;
typedef __attribute__((ext_vector_type(8)))  float  fvec8;
typedef __attribute__((ext_vector_type(4)))  u32    u32x4;

union Frag { bfvec16 v; u32x4 q[2]; };

static inline long cdivl(long a, long b) { return (a + b - 1) / b; }
static inline int ceil32(int c) { return (c + 31) & ~31; }

// ---------------------------------------------------------------------------
// Unified sparse/dense conv:  out[nout,co] = sum_k gather(Fb, nbr[k]) @ W[k]
// Fb : bf16 (n_in+1) x cip   (row n_in is all-zero; cip = ceil32(ci))
// Wt : bf16 K x co x cip     (transposed weights, zero-padded in ci)
// nbr: int   K x nout  (values in [0, n_in], n_in == missing)  | nullptr => identity
//
// Each wave computes a 32x32 output tile as 2x2 WMMA tiles (4 accumulators),
// reusing every A/B fragment across two v_wmma_f32_16x16x32_bf16 ops.
// 4 waves per block arranged 2(M) x 2(N) -> 64x64 tile per block.
// ---------------------------------------------------------------------------
__global__ __launch_bounds__(128) void spconv_wmma_kernel(
    const bf16_t* __restrict__ Fb, const bf16_t* __restrict__ Wt,
    const int* __restrict__ nbr, float* __restrict__ out,
    int n_out, int n_in, int cip, int co, int K)
{
  const int lane = threadIdx.x;              // 0..31 (wave32)
  const int wy   = threadIdx.y >> 1;         // wave's M sub-tile (0..1)
  const int wx   = threadIdx.y & 1;          // wave's N sub-tile (0..1)
  const int half = lane >> 4;                // 0 | 1
  const int l15  = lane & 15;

  const int rbase = (blockIdx.x * 2 + wy) * 32;   // first row of this wave
  const int cbase = (blockIdx.y * 2 + wx) * 32;   // first col of this wave
  if (cbase >= co) return;                        // wave-uniform exit

  const int mrow0 = rbase + l15;             // A rows gathered by this lane
  const int mrow1 = rbase + 16 + l15;
  const int ncol0 = cbase + l15;             // B cols loaded by this lane
  const int ncol1 = cbase + 16 + l15;
  const bool c0ok = (ncol0 < co);
  const bool c1ok = (ncol1 < co);

  fvec8 acc00 = {0.f,0.f,0.f,0.f,0.f,0.f,0.f,0.f};
  fvec8 acc01 = acc00, acc10 = acc00, acc11 = acc00;

  for (int k = 0; k < K; ++k) {
    int row0 = n_in, row1 = n_in;            // default: zero pad row
    if (mrow0 < n_out) {
      row0 = nbr ? nbr[(long)k * n_out + mrow0] : mrow0;
      if (row0 < 0 || row0 > n_in) row0 = n_in;
    }
    if (mrow1 < n_out) {
      row1 = nbr ? nbr[(long)k * n_out + mrow1] : mrow1;
      if (row1 < 0 || row1 > n_in) row1 = n_in;
    }
    const bf16_t* A0 = Fb + (long)row0 * cip;
    const bf16_t* A1 = Fb + (long)row1 * cip;
    const bf16_t* B0 = Wt + ((long)k * co + (c0ok ? ncol0 : 0)) * cip;
    const bf16_t* B1 = Wt + ((long)k * co + (c1ok ? ncol1 : 0)) * cip;

    for (int kk = 0; kk < cip; kk += 32) {
      Frag a0, a1, b0, b1;
      // ISA 16-bit A 16x32 layout: lanes 0-15 hold K={kk..kk+7, kk+16..kk+23},
      // lanes 16-31 hold K={kk+8..kk+15, kk+24..kk+31} (row-major contiguous)
      a0.q[0] = *(const u32x4*)(A0 + kk + half * 8);
      a0.q[1] = *(const u32x4*)(A0 + kk + 16 + half * 8);
      a1.q[0] = *(const u32x4*)(A1 + kk + half * 8);
      a1.q[1] = *(const u32x4*)(A1 + kk + 16 + half * 8);
      // B 32x16: lane<16 = col N, K {kk..kk+7, kk+16..kk+23} ; lane>=16 = col N, K {kk+8..kk+15, kk+24..kk+31}
      if (c0ok) {
        b0.q[0] = *(const u32x4*)(B0 + kk + half * 8);
        b0.q[1] = *(const u32x4*)(B0 + kk + 16 + half * 8);
      } else {
        u32x4 z = {0u,0u,0u,0u};
        b0.q[0] = z; b0.q[1] = z;
      }
      if (c1ok) {
        b1.q[0] = *(const u32x4*)(B1 + kk + half * 8);
        b1.q[1] = *(const u32x4*)(B1 + kk + 16 + half * 8);
      } else {
        u32x4 z = {0u,0u,0u,0u};
        b1.q[0] = z; b1.q[1] = z;
      }
      // 4 independent WMMAs: each fragment feeds two matrix ops
      acc00 = __builtin_amdgcn_wmma_f32_16x16x32_bf16(false, a0.v, false, b0.v,
                                                      (short)0, acc00, false, false);
      acc01 = __builtin_amdgcn_wmma_f32_16x16x32_bf16(false, a0.v, false, b1.v,
                                                      (short)0, acc01, false, false);
      acc10 = __builtin_amdgcn_wmma_f32_16x16x32_bf16(false, a1.v, false, b0.v,
                                                      (short)0, acc10, false, false);
      acc11 = __builtin_amdgcn_wmma_f32_16x16x32_bf16(false, a1.v, false, b1.v,
                                                      (short)0, acc11, false, false);
    }
  }

  // C/D layout: VGPR v -> M = v + half*8, N = l15
  #pragma unroll
  for (int vv = 0; vv < 8; ++vv) {
    int m0 = rbase + vv + half * 8;
    int m1 = rbase + 16 + vv + half * 8;
    if (m0 < n_out) {
      if (c0ok) out[(long)m0 * co + ncol0] = acc00[vv];
      if (c1ok) out[(long)m0 * co + ncol1] = acc01[vv];
    }
    if (m1 < n_out) {
      if (c0ok) out[(long)m1 * co + ncol0] = acc10[vv];
      if (c1ok) out[(long)m1 * co + ncol1] = acc11[vv];
    }
  }
}

// ---- transpose + bf16-convert weights:  W[K][ci][co] -> Wt[K][co][cip] ------
__global__ void wt_prep_kernel(const float* __restrict__ w, bf16_t* __restrict__ wt,
                               int K, int ci, int co, int cip, long total)
{
  long idx = (long)blockIdx.x * blockDim.x + threadIdx.x;
  if (idx >= total) return;
  long per = (long)co * cip;
  int  k   = (int)(idx / per);
  long rem = idx - (long)k * per;
  int  o   = (int)(rem / cip);
  int  i   = (int)(rem - (long)o * cip);
  float v  = (i < ci) ? w[((long)k * ci + i) * co + o] : 0.f;
  wt[idx] = (bf16_t)v;
}

// ---- per-channel mean / rsqrt(var+eps) over n rows --------------------------
__global__ void colstats_kernel(const float* __restrict__ x, int n, int c,
                                float* __restrict__ mean, float* __restrict__ inv)
{
  int ch = blockIdx.x;
  float s = 0.f, s2 = 0.f;
  for (int i = threadIdx.x; i < n; i += blockDim.x) {
    float v = x[(long)i * c + ch];
    s += v; s2 += v * v;
  }
  __shared__ float sh[256], sh2[256];
  sh[threadIdx.x] = s; sh2[threadIdx.x] = s2;
  __syncthreads();
  for (int st = blockDim.x >> 1; st > 0; st >>= 1) {
    if ((int)threadIdx.x < st) { sh[threadIdx.x] += sh[threadIdx.x + st];
                                 sh2[threadIdx.x] += sh2[threadIdx.x + st]; }
    __syncthreads();
  }
  if (threadIdx.x == 0) {
    float mu  = sh[0] / (float)n;
    float var = sh2[0] / (float)n - mu * mu;
    if (var < 0.f) var = 0.f;
    mean[ch] = mu;
    inv[ch]  = rsqrtf(var + 1e-5f);
  }
}

// ---- BN apply (+ optional residual add, relu); emits f32 and/or padded bf16 -
__global__ void bn_apply_kernel(const float* __restrict__ x,
                                const float* __restrict__ mean, const float* __restrict__ inv,
                                const float* __restrict__ g, const float* __restrict__ b,
                                const float* __restrict__ addsrc, int relu,
                                float* __restrict__ outf, bf16_t* __restrict__ outbf,
                                int n, int c, int cip, long total)
{
  long idx = (long)blockIdx.x * blockDim.x + threadIdx.x;
  if (idx >= total) return;
  int row = (int)(idx / cip);
  int col = (int)(idx - (long)row * cip);
  float y = 0.f;
  if (row < n && col < c) {
    float v = x[(long)row * c + col];
    if (mean) v = (v - mean[col]) * inv[col] * g[col] + b[col];
    if (addsrc) v += addsrc[(long)row * c + col];
    if (relu && v < 0.f) v = 0.f;
    y = v;
    if (outf) outf[(long)row * c + col] = y;
  }
  if (outbf) outbf[idx] = (bf16_t)y;
}

// ---- initial point->voxel segment sum ---------------------------------------
__global__ void p2v_kernel(const float* __restrict__ feats, const int* __restrict__ pt2vox,
                           float* __restrict__ acc, int npts, int c)
{
  long idx = (long)blockIdx.x * blockDim.x + threadIdx.x;
  if (idx >= (long)npts * c) return;
  int p = (int)(idx / c), ch = (int)(idx - (long)p * c);
  atomicAdd(&acc[(long)pt2vox[p] * c + ch], feats[idx]);
}

__global__ void div_rows_kernel(float* __restrict__ x, const float* __restrict__ counts,
                                int n, int c)
{
  long idx = (long)blockIdx.x * blockDim.x + threadIdx.x;
  if (idx >= (long)n * c) return;
  int row = (int)(idx / c);
  x[idx] /= counts[row];
}

// ---- trilinear devoxelize fused with point->voxel ---------------------------
__global__ void devox_p2v_kernel(const float* __restrict__ x0, const float* __restrict__ tri_w,
                                 const int* __restrict__ tri_idx, const int* __restrict__ pt2vox,
                                 float* __restrict__ acc, int npts, int c, int n0)
{
  long idx = (long)blockIdx.x * blockDim.x + threadIdx.x;
  if (idx >= (long)npts * c) return;
  int p = (int)(idx / c), ch = (int)(idx - (long)p * c);
  float z = 0.f;
  #pragma unroll
  for (int k = 0; k < 8; ++k) {
    int t = tri_idx[(long)k * npts + p];
    if (t < n0) z += tri_w[(long)k * npts + p] * x0[(long)t * c + ch];
  }
  atomicAdd(&acc[(long)pt2vox[p] * c + ch], z);
}

// ---- global pooling by batch index ------------------------------------------
__global__ void pool_kernel(const float* __restrict__ x, const int* __restrict__ b3,
                            float* __restrict__ pooled, float* __restrict__ cnt,
                            int n3, int c)
{
  long idx = (long)blockIdx.x * blockDim.x + threadIdx.x;
  if (idx >= (long)n3 * c) return;
  int row = (int)(idx / c), ch = (int)(idx - (long)row * c);
  int bi = b3[row];
  atomicAdd(&pooled[(long)bi * c + ch], x[idx]);
  if (ch == 0) atomicAdd(&cnt[bi], 1.f);
}

// ---- classifier + BN over batch dim (B=2, 40 classes) -----------------------
__global__ void cls_kernel(const float* __restrict__ pooled, const float* __restrict__ cnt,
                           const float* __restrict__ w, const float* __restrict__ bias,
                           const float* __restrict__ g, const float* __restrict__ b2,
                           float* __restrict__ out, int Bn, int c, int nc)
{
  __shared__ float o[256];
  int t = threadIdx.x;
  int bi = t / nc, cj = t - bi * nc;
  if (t < Bn * nc) {
    float s = bias[cj];
    const float* pr = pooled + (long)bi * c;
    float invc = 1.f / cnt[bi];
    for (int i = 0; i < c; ++i) s += (pr[i] * invc) * w[(long)i * nc + cj];
    o[bi * nc + cj] = s;
  }
  __syncthreads();
  if (t < Bn * nc) {
    float mu = 0.f;
    for (int i = 0; i < Bn; ++i) mu += o[i * nc + cj];
    mu /= (float)Bn;
    float var = 0.f;
    for (int i = 0; i < Bn; ++i) { float d = o[i * nc + cj] - mu; var += d * d; }
    var /= (float)Bn;
    out[bi * nc + cj] = (o[bi * nc + cj] - mu) * rsqrtf(var + 1e-5f) * g[cj] + b2[cj];
  }
}

// ===========================================================================
extern "C" void kernel_launch(void* const* d_in, const int* in_sizes, int n_in,
                              void* d_out, int out_size, void* d_ws, size_t ws_size,
                              hipStream_t stream)
{
  (void)out_size; (void)ws_size;
  const int CS[5]    = {64, 256, 512, 1024, 2048};
  const int MIDCS[4] = {64, 128, 256, 512};
  const int NBLK[4]  = {3, 8, 36, 3};
  const int Bn = 2, NC = 40, IN_DIM = 4;

  // ---- trailing 16 tensors (index/weight maps), located from the end -------
  int base = n_in - 16;
  const float* tri_w   = (const float*)d_in[base + 0];
  const int*   pt2vox  = (const int*)  d_in[base + 1];
  const float* counts0 = (const float*)d_in[base + 2];
  const int*   tri_idx = (const int*)  d_in[base + 3];
  const int*   b3      = (const int*)  d_in[base + 4];
  const int*   k3_v0   = (const int*)  d_in[base + 5];
  const int*   k2s1_v0 = (const int*)  d_in[base + 6];
  const int*   k3_d01  = (const int*)  d_in[base + 7];
  const int*   k2_d01  = (const int*)  d_in[base + 8];
  const int*   k3_v1   = (const int*)  d_in[base + 9];
  const int*   k3_d12  = (const int*)  d_in[base + 10];
  const int*   k2_d12  = (const int*)  d_in[base + 11];
  const int*   k3_v2   = (const int*)  d_in[base + 12];
  const int*   k3_d23  = (const int*)  d_in[base + 13];
  const int*   k2_d23  = (const int*)  d_in[base + 14];
  const int*   k3_v3   = (const int*)  d_in[base + 15];

  const int npts = in_sizes[0] / IN_DIM;
  const int n0   = in_sizes[base + 2];
  const int n1   = in_sizes[base + 9]  / 27;
  const int n2   = in_sizes[base + 12] / 27;
  const int n3   = in_sizes[base + 4];

  const float* feats = (const float*)d_in[0];

  // ---- walk params in setup_inputs() insertion order -----------------------
  struct BP { const float *w1,*g1,*b1,*w2,*g2,*b2,*w3,*g3,*b3,*wd,*gd,*bd; };
  static BP blocks[4][36];
  int pi = 1;
  const float* stem_w = (const float*)d_in[pi++];
  const float* stem_g = (const float*)d_in[pi++];
  const float* stem_b = (const float*)d_in[pi++];
  for (int s = 0; s < 4; ++s) {
    for (int bk = 0; bk < NBLK[s]; ++bk) {
      BP& p = blocks[s][bk];
      p.w1 = (const float*)d_in[pi++]; p.g1 = (const float*)d_in[pi++]; p.b1 = (const float*)d_in[pi++];
      p.w2 = (const float*)d_in[pi++]; p.g2 = (const float*)d_in[pi++]; p.b2 = (const float*)d_in[pi++];
      p.w3 = (const float*)d_in[pi++]; p.g3 = (const float*)d_in[pi++]; p.b3 = (const float*)d_in[pi++];
      if (bk == 0) { p.wd = (const float*)d_in[pi++]; p.gd = (const float*)d_in[pi++]; p.bd = (const float*)d_in[pi++]; }
      else         { p.wd = nullptr; p.gd = nullptr; p.bd = nullptr; }
    }
  }
  const float* cls_w  = (const float*)d_in[pi++];
  const float* cls_b  = (const float*)d_in[pi++];
  const float* cls_g  = (const float*)d_in[pi++];
  const float* cls_b2 = (const float*)d_in[pi++];

  // ---- workspace carving ---------------------------------------------------
  long SZ = std::max({(long)n0 * 256, (long)n1 * 512, (long)n2 * 1024,
                      (long)n3 * 2048, (long)n0 * 64});
  long BFSZ = std::max({(long)(n0 + 1) * 256, (long)(n1 + 1) * 512,
                        (long)(n2 + 1) * 1024, (long)(n3 + 1) * 2048,
                        (long)(n0 + 1) * 64});
  long WTSZ = (long)8 * 2048 * 1024;  // largest transposed weight (s4 wd)

  size_t off = 0;
  auto carve = [&](size_t bytes) -> void* {
    void* p = (char*)d_ws + off;
    off += (bytes + 255) & ~(size_t)255;
    return p;
  };
  float*  Fcur  = (float*)carve((size_t)SZ * 4);
  float*  Fnext = (float*)carve((size_t)SZ * 4);
  float*  R1    = (float*)carve((size_t)SZ * 4);
  float*  R2    = (float*)carve((size_t)SZ * 4);
  float*  R3    = (float*)carve((size_t)SZ * 4);
  bf16_t* FBF   = (bf16_t*)carve((size_t)BFSZ * 2);
  bf16_t* FBF2  = (bf16_t*)carve((size_t)BFSZ * 2);
  bf16_t* HBF   = (bf16_t*)carve((size_t)BFSZ * 2);
  bf16_t* H2BF  = (bf16_t*)carve((size_t)BFSZ * 2);
  bf16_t* WT    = (bf16_t*)carve((size_t)WTSZ * 2);
  float*  meanA = (float*)carve(2048 * 4);
  float*  invA  = (float*)carve(2048 * 4);
  float*  meanB = (float*)carve(2048 * 4);
  float*  invB  = (float*)carve(2048 * 4);
  float*  pooled = (float*)carve((size_t)(Bn * 2048 + 16) * 4);
  float*  cnt    = pooled + (size_t)Bn * 2048;

  // ---- launch helpers ------------------------------------------------------
  auto conv = [&](const bf16_t* Fb, const bf16_t* Wt, const int* nbr, float* out,
                  int nout, int nin_, int cip, int co, int K) {
    dim3 grid((unsigned)cdivl(nout, 64), (unsigned)cdivl(co, 64));
    dim3 blk(32, 4);
    spconv_wmma_kernel<<<grid, blk, 0, stream>>>(Fb, Wt, nbr, out, nout, nin_, cip, co, K);
  };
  auto wtprep = [&](const float* w, int K, int ci, int co, int cip) {
    long total = (long)K * co * cip;
    wt_prep_kernel<<<(unsigned)cdivl(total, 256), 256, 0, stream>>>(w, WT, K, ci, co, cip, total);
  };
  auto stats = [&](const float* x, int n, int c, float* mu, float* iv) {
    colstats_kernel<<<(unsigned)c, 256, 0, stream>>>(x, n, c, mu, iv);
  };
  auto bnapply = [&](const float* x, const float* mu, const float* iv,
                     const float* g, const float* b, const float* add, int relu,
                     float* outf, bf16_t* outbf, int n, int c, int cip) {
    long total = outbf ? (long)(n + 1) * cip : (long)n * cip;
    bn_apply_kernel<<<(unsigned)cdivl(total, 256), 256, 0, stream>>>(
        x, mu, iv, g, b, add, relu, outf, outbf, n, c, cip, total);
  };

  // ---- stem: v = p2v(feats); x0 = bn(sconv(v, stem_w, k3_v0)) --------------
  hipMemsetAsync(R2, 0, (size_t)n0 * IN_DIM * sizeof(float), stream);
  p2v_kernel<<<(unsigned)cdivl((long)npts * IN_DIM, 256), 256, 0, stream>>>(feats, pt2vox, R2, npts, IN_DIM);
  div_rows_kernel<<<(unsigned)cdivl((long)n0 * IN_DIM, 256), 256, 0, stream>>>(R2, counts0, n0, IN_DIM);
  bnapply(R2, nullptr, nullptr, nullptr, nullptr, nullptr, 0, nullptr, FBF, n0, IN_DIM, 32);
  wtprep(stem_w, 27, IN_DIM, 64, 32);
  conv(FBF, WT, k3_v0, R1, n0, n0, 32, 64, 27);
  stats(R1, n0, 64, meanA, invA);
  bnapply(R1, meanA, invA, stem_g, stem_b, nullptr, 0, R3, nullptr, n0, 64, 64); // x0 f32

  // ---- devoxelize + p2v -> Fcur (n0 x 64) ----------------------------------
  hipMemsetAsync(Fcur, 0, (size_t)n0 * 64 * sizeof(float), stream);
  devox_p2v_kernel<<<(unsigned)cdivl((long)npts * 64, 256), 256, 0, stream>>>(
      R3, tri_w, tri_idx, pt2vox, Fcur, npts, 64, n0);
  div_rows_kernel<<<(unsigned)cdivl((long)n0 * 64, 256), 256, 0, stream>>>(Fcur, counts0, n0, 64);
  bnapply(Fcur, nullptr, nullptr, nullptr, nullptr, nullptr, 0, nullptr, FBF, n0, 64, 64);

  // ---- residual stages -----------------------------------------------------
  const int* dM[4]  = {k3_v0, k3_d01, k3_d12, k3_d23};
  const int* dD[4]  = {k2s1_v0, k2_d01, k2_d12, k2_d23};
  const int* vMp[4] = {k3_v0, k3_v1, k3_v2, k3_v3};
  int nin_s[4]  = {n0, n0, n1, n2};
  int nout_s[4] = {n0, n1, n2, n3};

  auto runBlock = [&](const BP& p, int nin_, int nout_, int ci, int co, int cm,
                      const int* nbrM, const int* nbrD) {
    int cip_ci = ceil32(ci), cip_cm = ceil32(cm), cip_co = ceil32(co);
    // h = relu(bn(F @ w1))
    wtprep(p.w1, 1, ci, cm, cip_ci);
    conv(FBF, WT, nullptr, R1, nin_, nin_, cip_ci, cm, 1);
    stats(R1, nin_, cm, meanA, invA);
    bnapply(R1, meanA, invA, p.g1, p.b1, nullptr, 1, nullptr, HBF, nin_, cm, cip_cm);
    // h = relu(bn(sconv(h, w2, nbrM)))
    wtprep(p.w2, 27, cm, cm, cip_cm);
    conv(HBF, WT, nbrM, R2, nout_, nin_, cip_cm, cm, 27);
    stats(R2, nout_, cm, meanA, invA);
    bnapply(R2, meanA, invA, p.g2, p.b2, nullptr, 1, nullptr, H2BF, nout_, cm, cip_cm);
    // h = bn(h @ w3)
    wtprep(p.w3, 1, cm, co, cip_cm);
    conv(H2BF, WT, nullptr, R3, nout_, nout_, cip_cm, co, 1);
    stats(R3, nout_, co, meanA, invA);
    // shortcut
    const float* scsrc;
    if (p.wd) {
      wtprep(p.wd, 8, ci, co, cip_ci);
      conv(FBF, WT, nbrD, R1, nout_, nin_, cip_ci, co, 8);
      stats(R1, nout_, co, meanB, invB);
      bnapply(R1, meanB, invB, p.gd, p.bd, nullptr, 0, R2, nullptr, nout_, co, co);
      scsrc = R2;
    } else {
      scsrc = Fcur; // identity (ci==co, nin_==nout_)
    }
    // out = relu(h + sc) -> f32 Fnext + bf16 FBF2
    bnapply(R3, meanA, invA, p.g3, p.b3, scsrc, 1, Fnext, FBF2, nout_, co, cip_co);
    std::swap(Fcur, Fnext);
    std::swap(FBF, FBF2);
  };

  for (int s = 0; s < 4; ++s) {
    runBlock(blocks[s][0], nin_s[s], nout_s[s], CS[s], CS[s + 1], MIDCS[s], dM[s], dD[s]);
    for (int bk = 1; bk < NBLK[s]; ++bk)
      runBlock(blocks[s][bk], nout_s[s], nout_s[s], CS[s + 1], CS[s + 1], MIDCS[s],
               vMp[s], nullptr);
  }

  // ---- pooling + classifier + batch BN -------------------------------------
  hipMemsetAsync(pooled, 0, (size_t)(Bn * 2048 + 16) * sizeof(float), stream);
  pool_kernel<<<(unsigned)cdivl((long)n3 * 2048, 256), 256, 0, stream>>>(Fcur, b3, pooled, cnt, n3, 2048);
  cls_kernel<<<1, 128, 0, stream>>>(pooled, cnt, cls_w, cls_b, cls_g, cls_b2,
                                    (float*)d_out, Bn, 2048, NC);
}
